// Inducer_75788992905419
// MI455X (gfx1250) — matrix operand order repaired and checked by reading, not compiled, not run on hardware
//
#include <hip/hip_runtime.h>

// ---------------------------------------------------------------------------
// MI455X (gfx1250) 2-layer LSTMP + heads.
//  * bf16 WMMA 16x16x32 (f32 accum) everywhere
//  * input GEMMs: per-WG LDS staging of the shared B block, double buffered,
//    moved by the Tensor Data Mover (tensor_load_to_lds + s_wait_tensorcnt)
//    when the builtin exists, cooperative copy otherwise
//  * persistent per-layer scan: c-state in registers (WMMA C/D layout),
//    h broadcast through L2, split-K projection via global f32 atomics,
//    atomic grid barrier with s_sleep backoff
// ---------------------------------------------------------------------------

typedef __bf16 bf16_t;
typedef __attribute__((ext_vector_type(16))) __bf16 v16bf;
typedef __attribute__((ext_vector_type(8)))  __bf16 v8bf;
typedef __attribute__((ext_vector_type(8)))  float  v8f;

#define TSTEPS 256
#define BATCH  16
#define HID    4096
#define PROJ   512
#define G4     16384   // 4*HID
#define NWG    32      // persistent workgroups for the scan kernels

#if defined(__has_builtin)
#  if __has_builtin(__builtin_amdgcn_tensor_load_to_lds)
#    define HAS_TDM 1
#  endif
#endif
#ifndef HAS_TDM
#  define HAS_TDM 0
#endif

// -------------------------- small device helpers ---------------------------

__device__ __forceinline__ v8f vzero8() {
  v8f v;
#pragma unroll
  for (int i = 0; i < 8; ++i) v[i] = 0.0f;
  return v;
}

__device__ __forceinline__ v8f wmma_bf16(v16bf a, v16bf b, v8f c) {
  // D = A(16x32 bf16) * B(32x16 bf16) + C(16x16 f32)
  return __builtin_amdgcn_wmma_f32_16x16x32_bf16(
      /*neg_a=*/false, a, /*neg_b=*/false, b,
      /*c_mod=*/(short)0, c, /*reuse_a=*/false, /*reuse_b=*/false);
}

__device__ __forceinline__ v16bf combine16(v8bf lo, v8bf hi) {
  v16bf r;
#pragma unroll
  for (int i = 0; i < 8; ++i) { r[i] = lo[i]; r[i + 8] = hi[i]; }
  return r;
}

// A-fragment (16-bit A 16x32): lane row = lane&15,
// elems e -> K = kb + hi16*16 + e (16 contiguous, matches B)
__device__ __forceinline__ v16bf load_a_tile(const bf16_t* row, int kb, int hi16) {
  v8bf lo = *(const v8bf*)(row + kb + hi16 * 16);
  v8bf hi = *(const v8bf*)(row + kb + hi16 * 16 + 8);
  return combine16(lo, hi);
}

// B-fragment: column = lane&15, elems e -> K = kb + hi16*16 + e (16 contiguous)
__device__ __forceinline__ v16bf load_b_tile(const bf16_t* p) {
  v8bf lo = *(const v8bf*)(p);
  v8bf hi = *(const v8bf*)(p + 8);
  return combine16(lo, hi);
}

__device__ __forceinline__ float sigm(float x) { return 1.0f / (1.0f + __expf(-x)); }

// Monotonic atomic grid barrier (all NWG workgroups resident).
__device__ __forceinline__ void grid_barrier(unsigned* cnt, unsigned target) {
  __syncthreads();
  if (threadIdx.x == 0) {
    __threadfence();
    atomicAdd(cnt, 1u);
    while (atomicAdd(cnt, 0u) < target) __builtin_amdgcn_s_sleep(2);
    __threadfence();
  }
  __syncthreads();
}

#if HAS_TDM
typedef __attribute__((ext_vector_type(4))) unsigned int u32x4;
typedef __attribute__((ext_vector_type(8))) int i32x8;
typedef __attribute__((ext_vector_type(4))) int i32x4;

// TDM 2-D tile load: rows x 32 bf16 elements, row stride = K elements,
// packed contiguously into LDS at lds_off. D# per CDNA5 ISA ch.8.
__device__ __forceinline__ void tdm_load_2d(const bf16_t* gsrc, unsigned lds_off,
                                            int K, int rows) {
  unsigned long long ga = (unsigned long long)gsrc;
  u32x4 g0;
  g0[0] = 1u;                                            // count=1, user mode
  g0[1] = lds_off;                                       // lds_addr (bytes)
  g0[2] = (unsigned)ga;                                  // global_addr[31:0]
  g0[3] = (unsigned)((ga >> 32) & 0x01FFFFFFull)         // global_addr[56:32]
          | (2u << 30);                                  // type = 2 ("image")
  i32x8 g1;
  g1[0] = 0x00010000;                                    // data_size = 2 bytes
  g1[1] = (K & 0xFFFF) << 16;                            // tensor_dim0 lo16
  g1[2] = (int)(((unsigned)(K >> 16) & 0xFFFFu)          // tensor_dim0 hi16
                | (16384u << 16));                       // tensor_dim1 lo16
  g1[3] = 32 << 16;                                      // tile_dim0 = 32
  g1[4] = rows;                                          // tile_dim1
  g1[5] = K;                                             // dim0_stride lo32
  g1[6] = 0;
  g1[7] = 0;
  i32x4 z4; z4[0] = 0; z4[1] = 0; z4[2] = 0; z4[3] = 0;  // groups 2/3 unused
  i32x8 z8;
#pragma unroll
  for (int i = 0; i < 8; ++i) z8[i] = 0;
  // 6-arg form (clang-23 / therock HIP headers)
  __builtin_amdgcn_tensor_load_to_lds(g0, g1, z4, z4, z8, 0);
}
#endif

// ------------------------- f32 -> bf16 convert/pad -------------------------

__global__ __launch_bounds__(256) void k_cvt_bf16(
    const float* __restrict__ src, long ss, long sc,
    bf16_t* __restrict__ dst, long ds_, long dc,
    long R, long kc, long kd) {
  long tot = R * kd;
  for (long i = (long)blockIdx.x * blockDim.x + threadIdx.x; i < tot;
       i += (long)gridDim.x * blockDim.x) {
    long r = i / kd, k = i % kd;
    float v = (k < kc) ? src[r * ss + sc + k] : 0.0f;
    dst[r * ds_ + dc + k] = (bf16_t)v;
  }
}

// ------------------- time-parallel input GEMM (bf16 WMMA) -------------------
// C[M][N] = A_bf[M][K] * W_bf[N][K]^T + bias[N]   (M=4096, N=16384)
// All 8 waves of a WG share one 128x32 B block per K-step -> stage it in LDS
// (TDM async copy when available), double buffered.

__global__ __launch_bounds__(256) void k_gemm_xg(
    const bf16_t* __restrict__ A, const bf16_t* __restrict__ W,
    const float* __restrict__ bias, float* __restrict__ C,
    int M, int N, int K) {
  __shared__ bf16_t lds_B[2][128][32];                 // 16 KB, double buffer

  const int wave = threadIdx.x >> 5;
  const int lane = threadIdx.x & 31;
  const int hi16 = lane >> 4;
  const int l = lane & 15;
  const int mbase = blockIdx.y * 128 + wave * 16;      // 16 M rows per wave
  const int nbase = blockIdx.x * 128;                  // 8 N tiles per wave
  const int nk = K / 32;

  auto stage = [&](int buf, int kb) {
#if HAS_TDM
    if (threadIdx.x < 32)
      tdm_load_2d(W + (long)nbase * K + kb,
                  (unsigned)(unsigned long long)&lds_B[buf][0][0], K, 128);
#else
    for (int chunk = threadIdx.x; chunk < 512; chunk += 256) {
      int row = chunk >> 2, c8 = (chunk & 3) * 8;
      *(v8bf*)&lds_B[buf][row][c8] =
          *(const v8bf*)(W + (long)(nbase + row) * K + kb + c8);
    }
#endif
  };

  v8f acc[8];
#pragma unroll
  for (int nt = 0; nt < 8; ++nt) acc[nt] = vzero8();

  const bf16_t* arow = A + (long)(mbase + l) * K;
  stage(0, 0);

  for (int kbi = 0; kbi < nk; ++kbi) {
    const int kb = kbi * 32;
    const int buf = kbi & 1;
    if (kbi + 1 < nk) stage(buf ^ 1, kb + 32);         // prefetch next block
#if HAS_TDM
    if (threadIdx.x < 32) {                            // TDM completes in order:
      if (kbi + 1 < nk) __builtin_amdgcn_s_wait_tensorcnt((short)1);
      else              __builtin_amdgcn_s_wait_tensorcnt((short)0);
    }
#endif
    __syncthreads();

    __builtin_prefetch(arow + kb + 64, 0, 3);
    v16bf a = load_a_tile(arow, kb, hi16);
    v16bf b[8];
#pragma unroll
    for (int nt = 0; nt < 8; ++nt)
      b[nt] = load_b_tile(&lds_B[buf][nt * 16 + l][hi16 * 16]);
#pragma unroll
    for (int nt = 0; nt < 8; ++nt) acc[nt] = wmma_bf16(a, b[nt], acc[nt]);

    __syncthreads();                                   // protect buf reuse
  }

#pragma unroll
  for (int nt = 0; nt < 8; ++nt) {
#pragma unroll
    for (int r = 0; r < 8; ++r) {
      int m = mbase + r + hi16 * 8;
      int n = nbase + nt * 16 + l;
      C[(long)m * N + n] = acc[nt][r] + bias[n];
    }
  }
}

// ------------------------- persistent LSTMP scan ---------------------------

struct LstmParams {
  const float*  xg;       // [T][16][16384] precomputed input gate contributions
  const bf16_t* a0_bf;    // layer2: out1_bf [T][16][512]; layer1: nullptr
  bf16_t*       h_bf;     // [16][512] current projected state (bf16, L2 shared)
  const bf16_t* Wg;       // [16384][K] bf16 recurrent (L2: cat(W_ih2p,W_hh2))
  const bf16_t* Whr;      // [512][4096] bf16 projection
  float*        hacc;     // [2][16][512] split-K accumulator (double buffer)
  float*        out_f32;  // L1: out1 [T][16][512]; L2: resid = out2+out1
  const float*  addend;   // L2: out1; L1: nullptr
  bf16_t*       out_bf;   // L1: out1_bf; L2: nullptr
  float*        h_final;  // d_out h_T [16][512]
  float*        c_final;  // d_out c_T [16][4096]
  unsigned*     barrier;
  int           K;        // 512 (layer1) or 1024 (layer2)
};

__global__ __launch_bounds__(256) void k_lstmp(LstmParams p) {
  const int wave = threadIdx.x >> 5;
  const int lane = threadIdx.x & 31;
  const int hi16 = lane >> 4;
  const int l = lane & 15;
  const int jb = blockIdx.x * 128 + wave * 16;      // this wave's 16 hidden units
  const unsigned gtid = blockIdx.x * blockDim.x + threadIdx.x;  // 0..8191

  __shared__ bf16_t lds_s[BATCH][128];              // s slice restage for proj A

  v8f cstate = vzero8();                            // c[m=r+8*hi16][j=jb+l]

  // init: zero h (h0=0) and both accumulator buffers
  p.h_bf[gtid] = (bf16_t)0.0f;
  p.hacc[gtid] = 0.0f;
  p.hacc[BATCH * PROJ + gtid] = 0.0f;
  unsigned ep = 1;
  grid_barrier(p.barrier, ep * gridDim.x);

  for (int t = 0; t < TSTEPS; ++t) {
    // ---- gates: z[g] = xg[t] + [a0_t | h] @ Wg^T  (four 16x16 tiles) ----
    v8f z[4];
#pragma unroll
    for (int g = 0; g < 4; ++g) z[g] = vzero8();

    for (int kb = 0; kb < p.K; kb += 32) {
      const bf16_t* abase;
      int krow;
      if (p.a0_bf != nullptr && kb < PROJ) {
        abase = p.a0_bf + (long)t * (BATCH * PROJ);  krow = kb;
      } else {
        abase = p.h_bf;  krow = (p.a0_bf != nullptr) ? kb - PROJ : kb;
      }
      v16bf a = load_a_tile(abase + (long)l * PROJ, krow, hi16);
      v16bf bg[4];
#pragma unroll
      for (int g = 0; g < 4; ++g)
        bg[g] = load_b_tile(p.Wg + (long)(g * HID + jb + l) * p.K + kb +
                            hi16 * 16);
#pragma unroll
      for (int g = 0; g < 4; ++g) z[g] = wmma_bf16(a, bg[g], z[g]);
    }

    const float* xgt = p.xg + (long)t * (BATCH * G4);
#pragma unroll
    for (int r = 0; r < 8; ++r) {
      int m = r + hi16 * 8;
      const float* xr = xgt + (long)m * G4 + jb + l;
      z[0][r] += xr[0];
      z[1][r] += xr[HID];
      z[2][r] += xr[2 * HID];
      z[3][r] += xr[3 * HID];
    }

    // ---- elementwise cell update ----
    v8f s;
#pragma unroll
    for (int r = 0; r < 8; ++r) {
      float ig = sigm(z[0][r]);
      float fg = sigm(z[1][r]);
      float gg = tanhf(z[2][r]);
      float og = sigm(z[3][r]);
      float cv = fg * cstate[r] + ig * gg;
      cstate[r] = cv;
      s[r] = og * tanhf(cv);
    }
#pragma unroll
    for (int r = 0; r < 8; ++r)
      lds_s[r + hi16 * 8][wave * 16 + l] = (bf16_t)s[r];
    __syncthreads();

    // ---- projection partial: s_slice[16x128] @ Whr[:,slice]^T -> [16x512] ----
    v8f hp[4];
#pragma unroll
    for (int nt = 0; nt < 4; ++nt) hp[nt] = vzero8();
#pragma unroll
    for (int kk = 0; kk < 4; ++kk) {
      int kb = kk * 32;
      v16bf a = load_a_tile(&lds_s[l][0], kb, hi16);   // ds_load_b128 x2
      v16bf bw[4];
#pragma unroll
      for (int nt = 0; nt < 4; ++nt) {
        int pc = wave * 64 + nt * 16 + l;
        bw[nt] = load_b_tile(p.Whr + (long)pc * HID + (long)blockIdx.x * 128 +
                             kb + hi16 * 16);
      }
#pragma unroll
      for (int nt = 0; nt < 4; ++nt) hp[nt] = wmma_bf16(a, bw[nt], hp[nt]);
    }
    float* hc = p.hacc + (t & 1) * (BATCH * PROJ);
#pragma unroll
    for (int nt = 0; nt < 4; ++nt)
#pragma unroll
      for (int r = 0; r < 8; ++r)
        atomicAdd(&hc[(r + hi16 * 8) * PROJ + wave * 64 + nt * 16 + l],
                  hp[nt][r]);

    ep++; grid_barrier(p.barrier, ep * gridDim.x);

    // ---- reduce/convert phase: 8192 threads <-> 16x512 h entries ----
    float hv = hc[gtid];
    hc[gtid] = 0.0f;                                  // ready for t+2
    p.h_bf[gtid] = (bf16_t)hv;
    float ov = hv + (p.addend ? p.addend[(long)t * (BATCH * PROJ) + gtid] : 0.0f);
    p.out_f32[(long)t * (BATCH * PROJ) + gtid] = ov;
    if (p.out_bf) p.out_bf[(long)t * (BATCH * PROJ) + gtid] = (bf16_t)hv;
    if (t == TSTEPS - 1) p.h_final[gtid] = hv;

    ep++; grid_barrier(p.barrier, ep * gridDim.x);
  }

  // final cell state, c[m][j]
#pragma unroll
  for (int r = 0; r < 8; ++r)
    p.c_final[(long)(r + hi16 * 8) * HID + jb + l] = cstate[r];
}

// ------------------------------- heads -------------------------------------
// Packed-token heads: block = t, wave per token, lanes 0..19 = outputs.

__global__ __launch_bounds__(256) void k_heads(
    const float* __restrict__ resid, const int* __restrict__ lens,
    const float* __restrict__ Wp, const float* __restrict__ bp,
    const float* __restrict__ Wa, const float* __restrict__ ba,
    float* __restrict__ pssm, float* __restrict__ aa) {
  int t = blockIdx.x;
  int cnt = 0; long off = 0;
  for (int b = 0; b < BATCH; ++b) {
    int L = lens[b];
    cnt += (L > t);
    off += (L < t ? L : t);   // offset_t = sum_b min(len_b, t)
  }
  int wave = threadIdx.x >> 5, lane = threadIdx.x & 31;
  for (int b = wave; b < cnt; b += 8) {
    const float* x = resid + ((long)t * BATCH + b) * PROJ;
    long row = off + b;
    float logit = -INFINITY;
    if (lane < 20) {
      const float* w = Wp + (long)lane * PROJ;
      float s = bp[lane];
      for (int k = 0; k < PROJ; ++k) s += x[k] * w[k];
      logit = s;
    }
    float mx = logit;
    for (int d = 16; d; d >>= 1) mx = fmaxf(mx, __shfl_xor(mx, d, 32));
    float e = (lane < 20) ? __expf(logit - mx) : 0.0f;
    float se = e;
    for (int d = 16; d; d >>= 1) se += __shfl_xor(se, d, 32);
    if (lane < 20) pssm[row * 20 + lane] = e / se;
    if (lane < 20) {
      const float* w = Wa + (long)lane * PROJ;
      float s = ba[lane];
      for (int k = 0; k < PROJ; ++k) s += x[k] * w[k];
      aa[row * 20 + lane] = s;
    }
  }
}

// ------------------------------ launcher -----------------------------------

extern "C" void kernel_launch(void* const* d_in, const int* in_sizes, int n_in,
                              void* d_out, int out_size, void* d_ws, size_t ws_size,
                              hipStream_t stream) {
  (void)in_sizes; (void)n_in; (void)ws_size;
  const float* uncon  = (const float*)d_in[0];
  const float* conx   = (const float*)d_in[1];
  const int*   lens   = (const int*)d_in[2];
  const float* W_ih1  = (const float*)d_in[3];
  const float* b1     = (const float*)d_in[4];
  const float* W_hh1  = (const float*)d_in[5];
  const float* W_hr1  = (const float*)d_in[6];
  const float* W_ih2  = (const float*)d_in[7];
  const float* b2     = (const float*)d_in[8];
  const float* W_hh2  = (const float*)d_in[9];
  const float* W_hr2  = (const float*)d_in[10];
  const float* W_pssm = (const float*)d_in[11];
  const float* b_pssm = (const float*)d_in[12];
  const float* W_aa   = (const float*)d_in[13];
  const float* b_aa   = (const float*)d_in[14];

  const long MROWS = (long)TSTEPS * BATCH;   // 4096
  const long K1P = 544;                      // 532 padded to 17*32
  const long KC  = 768;                      // con_x features

  char* base = (char*)d_ws;
  size_t off = 0;
  auto carve = [&](size_t bytes) -> void* {
    void* p = base + off;
    off = (off + bytes + 255) & ~(size_t)255;
    return p;
  };

  unsigned* bar1   = (unsigned*)carve(256);
  unsigned* bar2   = (unsigned*)carve(256);
  bf16_t* A1_bf    = (bf16_t*)carve(MROWS * K1P * 2);
  bf16_t* Wih1_bf  = (bf16_t*)carve((long)G4 * K1P * 2);
  bf16_t* Ac_bf    = (bf16_t*)carve(MROWS * KC * 2);
  bf16_t* Wih2c_bf = (bf16_t*)carve((long)G4 * KC * 2);
  bf16_t* Whh1_bf  = (bf16_t*)carve((long)G4 * PROJ * 2);
  bf16_t* W2cat_bf = (bf16_t*)carve((long)G4 * 1024 * 2);
  bf16_t* Whr1_bf  = (bf16_t*)carve((long)PROJ * HID * 2);
  bf16_t* Whr2_bf  = (bf16_t*)carve((long)PROJ * HID * 2);
  float*  xg1      = (float*)carve(MROWS * (long)G4 * 4);
  float*  xg2      = (float*)carve(MROWS * (long)G4 * 4);
  float*  out1     = (float*)carve(MROWS * (long)PROJ * 4);
  bf16_t* out1_bf  = (bf16_t*)carve(MROWS * (long)PROJ * 2);
  float*  resid    = (float*)carve(MROWS * (long)PROJ * 4);
  float*  hacc     = (float*)carve(2L * BATCH * PROJ * 4);
  bf16_t* h_bf     = (bf16_t*)carve((long)BATCH * PROJ * 2);

  // output layout: pssm[Ntok,20], aa[Ntok,20], h1[16,512], c1[16,4096],
  //                h2[16,512], c2[16,4096]
  long Ntok = ((long)out_size - 2L * BATCH * (PROJ + HID)) / 40;
  float* outp  = (float*)d_out;
  float* pssm  = outp;
  float* aa    = outp + Ntok * 20;
  float* h1o   = outp + Ntok * 40;
  float* c1o   = h1o + (long)BATCH * PROJ;
  float* h2o   = c1o + (long)BATCH * HID;
  float* c2o   = h2o + (long)BATCH * PROJ;

  (void)hipMemsetAsync(bar1, 0, 256, stream);
  (void)hipMemsetAsync(bar2, 0, 256, stream);

  auto cvt = [&](const float* s, long ss, long sc, bf16_t* d, long ds_, long dc,
                 long R, long kc, long kd) {
    long tot = R * kd;
    long blocks = (tot + 255) / 256;
    if (blocks > 8192) blocks = 8192;
    k_cvt_bf16<<<(int)blocks, 256, 0, stream>>>(s, ss, sc, d, ds_, dc, R, kc, kd);
  };

  cvt(uncon, 532, 0, A1_bf, K1P, 0, MROWS, 532, K1P);        // pad K 532->544
  cvt(W_ih1, 532, 0, Wih1_bf, K1P, 0, G4, 532, K1P);
  cvt(conx, KC, 0, Ac_bf, KC, 0, MROWS, KC, KC);
  cvt(W_ih2, 1280, 512, Wih2c_bf, KC, 0, G4, KC, KC);        // con part of W_ih2
  cvt(W_hh1, PROJ, 0, Whh1_bf, PROJ, 0, G4, PROJ, PROJ);
  cvt(W_ih2, 1280, 0, W2cat_bf, 1024, 0, G4, PROJ, PROJ);    // cat: [W_ih2p |
  cvt(W_hh2, PROJ, 0, W2cat_bf, 1024, PROJ, G4, PROJ, PROJ); //        W_hh2 ]
  cvt(W_hr1, HID, 0, Whr1_bf, HID, 0, PROJ, HID, HID);
  cvt(W_hr2, HID, 0, Whr2_bf, HID, 0, PROJ, HID, HID);

  // time-parallel input GEMMs
  dim3 gg(G4 / 128, (int)(MROWS / 128));
  k_gemm_xg<<<gg, 256, 0, stream>>>(A1_bf, Wih1_bf, b1, xg1,
                                    (int)MROWS, G4, (int)K1P);
  k_gemm_xg<<<gg, 256, 0, stream>>>(Ac_bf, Wih2c_bf, b2, xg2,
                                    (int)MROWS, G4, (int)KC);

  // layer 1 scan
  LstmParams p1;
  p1.xg = xg1;       p1.a0_bf = nullptr;  p1.h_bf = h_bf;
  p1.Wg = Whh1_bf;   p1.Whr = Whr1_bf;    p1.hacc = hacc;
  p1.out_f32 = out1; p1.addend = nullptr; p1.out_bf = out1_bf;
  p1.h_final = h1o;  p1.c_final = c1o;    p1.barrier = bar1; p1.K = PROJ;
  k_lstmp<<<NWG, 256, 0, stream>>>(p1);

  // layer 2 scan (input = [out1_t | h2], K=1024); writes resid = out2 + out1
  LstmParams p2;
  p2.xg = xg2;        p2.a0_bf = out1_bf; p2.h_bf = h_bf;
  p2.Wg = W2cat_bf;   p2.Whr = Whr2_bf;   p2.hacc = hacc;
  p2.out_f32 = resid; p2.addend = out1;   p2.out_bf = nullptr;
  p2.h_final = h2o;   p2.c_final = c2o;   p2.barrier = bar2; p2.K = 1024;
  k_lstmp<<<NWG, 256, 0, stream>>>(p2);

  // packed heads
  k_heads<<<TSTEPS, 256, 0, stream>>>(resid, lens, W_pssm, b_pssm, W_aa, b_aa,
                                      pssm, aa);
}
